// LSTMCell_5557687681205
// MI455X (gfx1250) — matrix operand order repaired and checked by reading, not compiled, not run on hardware
//
#include <hip/hip_runtime.h>
#include <hip/hip_bf16.h>

typedef _Float16 half16 __attribute__((ext_vector_type(16)));
typedef _Float16 half8  __attribute__((ext_vector_type(8)));
typedef _Float16 half4  __attribute__((ext_vector_type(4)));
typedef float    v8f    __attribute__((ext_vector_type(8)));
typedef float    f4v    __attribute__((ext_vector_type(4)));
typedef unsigned int u32x4 __attribute__((ext_vector_type(4)));
typedef int      i32x4  __attribute__((ext_vector_type(4)));
typedef int      i32x8  __attribute__((ext_vector_type(8)));

#define B_DIM 16384
#define I_DIM 512
#define H_DIM 512
#define K_TOT 1024          // concatenated K = I + H
#define KSTEP 32            // K per v_wmma_f32_16x16x32_f16
#define LDB   40            // padded LDS row (halfs): 80B stride, conflict-free b128

#if defined(__has_builtin) && __has_builtin(__builtin_amdgcn_sched_group_barrier)
#define SGB(m, n, id) __builtin_amdgcn_sched_group_barrier((m), (n), (id))
#else
#define SGB(m, n, id)
#endif

// ---------------- pack kernels: fp32 -> f16, concatenated K ----------------

__global__ void pack_A(const float* __restrict__ x, const float* __restrict__ z,
                       _Float16* __restrict__ A) {
  int t = blockIdx.x * blockDim.x + threadIdx.x;
  int base = t * 4;
  int b = base >> 10;
  int k = base & 1023;
  const float* src = (k < I_DIM) ? (x + b * I_DIM + k) : (z + b * H_DIM + (k - I_DIM));
  f4v v = *(const f4v*)src;
  half4 h = { (_Float16)v[0], (_Float16)v[1], (_Float16)v[2], (_Float16)v[3] };
  *(half4*)(A + base) = h;
}

__global__ void pack_W(const float* __restrict__ W0, const float* __restrict__ W1,
                       const float* __restrict__ W2, const float* __restrict__ W3,
                       const float* __restrict__ U0, const float* __restrict__ U1,
                       const float* __restrict__ U2, const float* __restrict__ U3,
                       _Float16* __restrict__ Wc) {
  int t = blockIdx.x * blockDim.x + threadIdx.x;
  int base = t * 4;
  int g = base >> 19;
  int r = (base >> 10) & 511;
  int k = base & 1023;
  const float* Wg = (g == 0) ? W0 : (g == 1) ? W1 : (g == 2) ? W2 : W3;
  const float* Ug = (g == 0) ? U0 : (g == 1) ? U1 : (g == 2) ? U2 : U3;
  const float* src = (k < I_DIM) ? (Wg + r * I_DIM + k) : (Ug + r * H_DIM + (k - I_DIM));
  f4v v = *(const f4v*)src;
  half4 h = { (_Float16)v[0], (_Float16)v[1], (_Float16)v[2], (_Float16)v[3] };
  *(half4*)(Wc + base) = h;
}

// ---------------- TDM descriptors (ISA §8.3-8.6) ----------------
// 16-bit elems, HW padding: 4 DWORDs after every 16 DWORDs stored
// (rows of 32 halfs -> LDS stride 40 halfs).

struct TdmGroups { u32x4 g0; i32x8 g1; i32x4 g2; i32x4 g3; };

__device__ __forceinline__ TdmGroups tdm_make(
    unsigned td0, unsigned td1, unsigned td2,       // tensor extents (elems)
    unsigned tile0, unsigned tile1, unsigned tile2, // tile extents
    unsigned stride0, unsigned long long stride1)   // row / plane strides (elems)
{
  TdmGroups d;
  d.g0[0] = 1u;                                      // count=1 (valid), load
  d.g0[1] = 0u;                                      // lds_addr (patched per issue)
  d.g0[2] = 0u;                                      // global_addr lo (patched)
  d.g0[3] = (2u << 30);                              // type=2 (patched with addr hi)
  d.g1[0] = (int)((1u << 16)      // data_size = 2 bytes
                | (1u << 20)      // pad_enable
                | (3u << 22)      // pad_interval: 16 DWORDs
                | (3u << 25));    // pad_amount:   4 DWORDs
  d.g1[1] = (int)((td0 & 0xFFFFu) << 16);
  d.g1[2] = (int)((td0 >> 16) | ((td1 & 0xFFFFu) << 16));
  d.g1[3] = (int)((td1 >> 16) | (tile0 << 16));
  d.g1[4] = (int)(tile1 | (tile2 << 16));
  d.g1[5] = (int)stride0;
  d.g1[6] = (int)((unsigned)((stride1 & 0xFFFFull) << 16));
  d.g1[7] = (int)((unsigned)(stride1 >> 16));
  d.g2[0] = (int)td2; d.g2[1] = 0; d.g2[2] = 0; d.g2[3] = 0;
  d.g3[0] = 0; d.g3[1] = 0; d.g3[2] = 0; d.g3[3] = 0;
  return d;
}

__device__ __forceinline__ void tdm_issue(const TdmGroups& d, const _Float16* gptr,
                                          unsigned lds_byte_off) {
  unsigned long long ga = (unsigned long long)(uintptr_t)gptr;
  u32x4 g0 = d.g0;
  g0[1] = lds_byte_off;
  g0[2] = (unsigned)ga;
  g0[3] = (unsigned)((ga >> 32) & 0x1FFFFFFu) | (2u << 30);
#if defined(__clang_major__) && __clang_major__ >= 23
  i32x8 g4; g4[0]=0; g4[1]=0; g4[2]=0; g4[3]=0; g4[4]=0; g4[5]=0; g4[6]=0; g4[7]=0;
  __builtin_amdgcn_tensor_load_to_lds(g0, d.g1, d.g2, d.g3, g4, 0);
#else
  __builtin_amdgcn_tensor_load_to_lds(g0, d.g1, d.g2, d.g3, 0);
#endif
}

// ---------------- fused 4-gate GEMM + LSTM pointwise ----------------

__device__ __forceinline__ float sigmoid_(float v) {
  return 1.0f / (1.0f + __expf(-v));
}
__device__ __forceinline__ float tanh_(float v) {
  v = fminf(15.0f, fmaxf(-15.0f, v));
  float e = __expf(2.0f * v);
  return (e - 1.0f) / (e + 1.0f);
}
__device__ __forceinline__ half16 cat8(half8 lo, half8 hi) {
  return __builtin_shufflevector(lo, hi, 0,1,2,3,4,5,6,7,8,9,10,11,12,13,14,15);
}

#define SA_STAGE_BYTES (64 * LDB * 2)    // 5120
#define SB_STAGE_BYTES (512 * LDB * 2)   // 40960

__global__ __launch_bounds__(512, 1)
void lstm_wmma(const _Float16* __restrict__ A,    // [B][1024] f16
               const _Float16* __restrict__ Wc,   // [4][512][1024] f16
               const float* __restrict__ b0, const float* __restrict__ b1,
               const float* __restrict__ b2, const float* __restrict__ b3,
               const float* __restrict__ z,       // [B][H] fp32
               float* __restrict__ out) {         // h_next then c_next
  extern __shared__ _Float16 smem[];
  _Float16* sA = smem;                   // [2][64][LDB]
  _Float16* sB = smem + 2 * 64 * LDB;    // [2][4*128][LDB]

  const int r0 = blockIdx.y * 64;        // batch-row tile
  const int c0 = blockIdx.x * 128;       // H-col tile
  const int tid  = threadIdx.x;
  const int lane = tid & 31;
  const int wave = tid >> 5;             // 0..15
  const int wm = wave & 3;               // wave M-tile (0..3)
  const int wn = wave >> 2;              // wave N-group (0..3), 32 cols each

  const unsigned lds_base = (unsigned)(uintptr_t)(void*)smem;
  const unsigned sA_off = lds_base;
  const unsigned sB_off = lds_base + 2 * SA_STAGE_BYTES;

  v8f acc[4][2];
  #pragma unroll
  for (int g = 0; g < 4; ++g)
    #pragma unroll
    for (int j = 0; j < 2; ++j)
      acc[g][j] = (v8f){0.f,0.f,0.f,0.f,0.f,0.f,0.f,0.f};

  // invariant descriptor parts (SGPR-resident, hoisted out of K loop)
  const TdmGroups descB = tdm_make(K_TOT, H_DIM, 4, KSTEP, 128, 4,
                                   K_TOT, (unsigned long long)H_DIM * K_TOT);
  const TdmGroups descA = tdm_make(K_TOT, B_DIM, 0, KSTEP, 64, 0,
                                   K_TOT, 0ull);
  const _Float16* gB = Wc + (size_t)c0 * K_TOT;   // gate-0 tile base
  const _Float16* gA = A + (size_t)r0 * K_TOT;

  auto issueStage = [&](int stg, int k0) {
    if (wave == 0)      tdm_issue(descB, gB + k0, sB_off + stg * SB_STAGE_BYTES);
    else if (wave == 1) tdm_issue(descA, gA + k0, sA_off + stg * SA_STAGE_BYTES);
  };

  // ISA §7.12.2 fragment addressing (wave32)
  const int mr = lane & 15;              // A: row within 16-row tile
  const int kh = (lane >> 4) * 8;        // A: K sub-offset (0 or 8)
  const int nl = lane & 15;              // B: col within 16-col tile
  const int kb = (lane >> 4) * 8;        // B: K sub-offset (0 or 8)

  issueStage(0, 0);
  if (wave < 2) __builtin_amdgcn_s_wait_tensorcnt(0);
  __syncthreads();

  const int nsteps = K_TOT / KSTEP;      // 32
  #pragma unroll 1
  for (int s = 0; s < nsteps; ++s) {
    const int cur = s & 1;
    if (s + 1 < nsteps) issueStage(cur ^ 1, (s + 1) * KSTEP);

    const _Float16* sAc = sA + cur * 64 * LDB;
    const _Float16* sBc = sB + cur * 512 * LDB;

    const _Float16* ap = sAc + (wm * 16 + mr) * LDB + kh;
    half16 a = cat8(*(const half8*)ap, *(const half8*)(ap + 16));

    #pragma unroll
    for (int g = 0; g < 4; ++g) {
      const _Float16* gb = sBc + (g * 128 + wn * 32 + nl) * LDB + kb;
      half16 bf0 = cat8(*(const half8*)(gb),             *(const half8*)(gb + 16));
      half16 bf1 = cat8(*(const half8*)(gb + 16 * LDB),  *(const half8*)(gb + 16 * LDB + 16));
      acc[g][0] = __builtin_amdgcn_wmma_f32_16x16x32_f16(false, a, false, bf0, (short)0, acc[g][0], false, false);
      acc[g][1] = __builtin_amdgcn_wmma_f32_16x16x32_f16(false, a, false, bf1, (short)0, acc[g][1], false, false);
    }

    // schedule shaping: keep 2 fragment-loads in flight ahead of each WMMA
    // per k-step: 18 ds_loads (A:2 + B:16), 8 WMMAs
    SGB(0x100, 6, 0);                 // A pair + 2 B fragments
    SGB(0x008, 1, 0); SGB(0x100, 2, 0);
    SGB(0x008, 1, 0); SGB(0x100, 2, 0);
    SGB(0x008, 1, 0); SGB(0x100, 2, 0);
    SGB(0x008, 1, 0); SGB(0x100, 2, 0);
    SGB(0x008, 1, 0); SGB(0x100, 2, 0);
    SGB(0x008, 1, 0); SGB(0x100, 2, 0);
    SGB(0x008, 2, 0);                 // drain last 2 WMMAs

    if (s + 1 < nsteps && wave < 2) __builtin_amdgcn_s_wait_tensorcnt(0);
    __syncthreads();
  }

  // epilogue: bias + gates, write h_next / c_next
  #pragma unroll
  for (int j = 0; j < 2; ++j) {
    const int nG = c0 + wn * 32 + j * 16 + nl;
    const float bi_ = b0[nG], bf_ = b1[nG], bc_ = b2[nG], bo_ = b3[nG];
    const int mBase = r0 + wm * 16 + (lane >> 4) * 8;
    #pragma unroll
    for (int r = 0; r < 8; ++r) {
      const int m = mBase + r;
      float iv = sigmoid_(acc[0][j][r] + bi_);
      float fv = sigmoid_(acc[1][j][r] + bf_);
      float gv = tanh_   (acc[2][j][r] + bc_);
      float ov = sigmoid_(acc[3][j][r] + bo_);
      float zv = z[m * H_DIM + nG];
      float cn = fv * zv + iv * gv;
      float hn = ov * tanh_(cn);
      out[(size_t)m * H_DIM + nG] = hn;
      out[(size_t)B_DIM * H_DIM + (size_t)m * H_DIM + nG] = cn;
    }
  }
}

// ---------------- host launch ----------------

extern "C" void kernel_launch(void* const* d_in, const int* in_sizes, int n_in,
                              void* d_out, int out_size, void* d_ws, size_t ws_size,
                              hipStream_t stream) {
  const float* z  = (const float*)d_in[0];
  const float* x  = (const float*)d_in[1];
  const float* Wi = (const float*)d_in[2];
  const float* Wf = (const float*)d_in[3];
  const float* Wcg= (const float*)d_in[4];
  const float* Wo = (const float*)d_in[5];
  const float* bi = (const float*)d_in[6];
  const float* bf = (const float*)d_in[7];
  const float* bc = (const float*)d_in[8];
  const float* bo = (const float*)d_in[9];
  const float* Ui = (const float*)d_in[10];
  const float* Uf = (const float*)d_in[11];
  const float* Uc = (const float*)d_in[12];
  const float* Uo = (const float*)d_in[13];
  float* out = (float*)d_out;

  _Float16* Acat = (_Float16*)d_ws;                                      // 33.5 MB
  _Float16* Wcat = (_Float16*)((char*)d_ws + (size_t)B_DIM * K_TOT * 2); // 4 MB

  pack_A<<<(B_DIM * K_TOT / 4) / 256, 256, 0, stream>>>(x, z, Acat);
  pack_W<<<(4 * H_DIM * K_TOT / 4) / 256, 256, 0, stream>>>(
      Wi, Wf, Wcg, Wo, Ui, Uf, Uc, Uo, Wcat);

  dim3 grid(H_DIM / 128, B_DIM / 64);   // (4, 256)
  size_t shmem = 2 * (size_t)(SA_STAGE_BYTES + SB_STAGE_BYTES); // 92160 B
  lstm_wmma<<<grid, 512, shmem, stream>>>(Acat, Wcat, bi, bf, bc, bo, z, out);
}
